// MoEMLP_13262859010707
// MI455X (gfx1250) — compile-verified, matrix operand order checked by reading, not run on hardware
//
#include <hip/hip_runtime.h>
#include <math.h>

typedef __attribute__((ext_vector_type(16))) __bf16 v16bf;
typedef __attribute__((ext_vector_type(8)))  float  v8f;

#define L_  2048
#define N_  8
#define M_  (L_*N_)      // 16384 tokens
#define E_  2048
#define H_  8192
#define MT  32           // tokens per workgroup
#define HC  128          // H-chunk held in LDS
#define NWAVE 16
#define TPB  (NWAVE*32)  // 512 threads = 16 wave32

#define XROW (E_+8)      // padded LDS row (elements) for x tile
#define HROW (HC+8)      // padded LDS row (elements) for h tile

// LDS layout (bytes)
#define LDS_B1   0                          // H_ floats      (32768 B)
#define LDS_B2   (LDS_B1 + H_*4)            // E_ floats      ( 8192 B)
#define LDS_X    (LDS_B2 + E_*4)            // MT*XROW bf16   (131584 B)
#define LDS_H    (LDS_X + MT*XROW*2)        // 2 * MT*HROW bf16 (17408 B)
#define LDS_TOTAL (LDS_H + 2*MT*HROW*2)     // 189952 B

union Frag { v16bf v; uint4 q[2]; };

static __device__ __forceinline__ unsigned short f2bf(float f) {
    unsigned u = __float_as_uint(f);
    u += 0x7FFFu + ((u >> 16) & 1u);          // round-to-nearest-even
    return (unsigned short)(u >> 16);
}

// Load a 16x32 bf16 fragment's per-lane data from LDS (two 16B chunks 32B apart).
static __device__ __forceinline__ v16bf ldfrag(const unsigned short* p) {
    Frag f;
    f.q[0] = *(const uint4*)(p);
    f.q[1] = *(const uint4*)(p + 16);
    return f.v;
}

static __device__ __forceinline__ v8f wmma_bf16(v16bf a, v16bf b, v8f c) {
    return __builtin_amdgcn_wmma_f32_16x16x32_bf16(
        false, a, false, b, (short)0, c, false, false);
}

static __device__ __forceinline__ v8f wmma_pair(v16bf a, const uint4& q0,
                                                const uint4& q1, v8f c) {
    Frag b; b.q[0] = q0; b.q[1] = q1;
    return wmma_bf16(a, b.v, c);
}

// ---- cluster-multicast weight loads (downgrade to global loads w/o cluster) ----
static __device__ __forceinline__ void cl_issue(const unsigned short* p,
                                                uint4& q0, uint4& q1) {
    asm volatile("s_mov_b32 m0, 0xffff\n\t"
                 "cluster_load_b128 %0, %2, off\n\t"
                 "cluster_load_b128 %1, %2, off offset:32"
                 : "=&v"(q0), "=&v"(q1)
                 : "v"(p)
                 : "memory");
}
// Fenced wait: sched_barrier(0) keeps the scheduler from hoisting the (pure)
// WMMA intrinsics above the counter wait.
template <int N>
static __device__ __forceinline__ void cl_wait() {
    __builtin_amdgcn_sched_barrier(0);
    asm volatile("s_wait_loadcnt %0" :: "i"(N) : "memory");
    __builtin_amdgcn_sched_barrier(0);
}
static __device__ __forceinline__ void cl_drain() {
    __builtin_amdgcn_sched_barrier(0);
    asm volatile("s_wait_loadcnt 0x0" ::: "memory");
    __builtin_amdgcn_sched_barrier(0);
}

// ---------- fp32 -> bf16 pre-pass ----------
extern "C" __global__ void moe_cvt_bf16(const float* __restrict__ s,
                                        unsigned short* __restrict__ d, int n4) {
    for (int i = blockIdx.x * blockDim.x + threadIdx.x; i < n4;
         i += gridDim.x * blockDim.x) {
        float4 v = ((const float4*)s)[i];
        unsigned lo = (unsigned)f2bf(v.x) | ((unsigned)f2bf(v.y) << 16);
        unsigned hi = (unsigned)f2bf(v.z) | ((unsigned)f2bf(v.w) << 16);
        ((uint2*)d)[i] = make_uint2(lo, hi);
    }
}

// ---------- fused 2-layer GELU MLP ----------
extern "C" __global__ __launch_bounds__(TPB)
void moe_mlp_fused(const unsigned short* __restrict__ xb,   // [M, E] bf16
                   const unsigned short* __restrict__ w1,   // [H, E] bf16
                   const unsigned short* __restrict__ w2,   // [E, H] bf16
                   const float* __restrict__ b1,
                   const float* __restrict__ b2,
                   float* __restrict__ out) {
    extern __shared__ char smem[];
    float*          ldsb1 = (float*)(smem + LDS_B1);
    float*          ldsb2 = (float*)(smem + LDS_B2);
    unsigned short* ldsx  = (unsigned short*)(smem + LDS_X);
    unsigned short* ldsh  = (unsigned short*)(smem + LDS_H);

    const int tid  = threadIdx.x;
    const int wave = tid >> 5;
    const int lane = tid & 31;
    const int half = lane >> 4;
    const int lrow = lane & 15;
    const int gm0  = blockIdx.x * MT;

    // ---- Phase 0: stage biases (f32) and x tile (bf16) into LDS ----
    for (int i = tid; i < H_; i += TPB) ldsb1[i] = b1[i];
    for (int i = tid; i < E_; i += TPB) ldsb2[i] = b2[i];
    for (int i = tid * 8; i < MT * E_; i += TPB * 8) {
        int row = i >> 11;          // / E_
        int col = i & (E_ - 1);
        *(uint4*)(ldsx + row * XROW + col) =
            *(const uint4*)(xb + (size_t)(gm0 + row) * E_ + col);
    }
    __syncthreads();

    // Persistent accumulators: wave owns out cols [wave*128, wave*128+128)
    v8f acc[2][8];
    #pragma unroll
    for (int a = 0; a < 2; ++a)
        #pragma unroll
        for (int b = 0; b < 8; ++b) acc[a][b] = {};

    const int mt1 = wave >> 3;
    const int nt1 = wave & 7;
    const int wcol0 = wave * 128;
    const unsigned short* arow1 = ldsx + (mt1 * 16 + lrow) * XROW + half * 8;

    for (int hc = 0; hc < H_; hc += HC) {
        unsigned short* hbuf = ldsh + ((hc >> 7) & 1) * (MT * HROW);

        // ---- GEMM1: one 16x16 h-tile per wave, K = E, pipelined cluster loads
        v8f c1 = {};
        const unsigned short* brow1 =
            w1 + (size_t)(hc + nt1 * 16 + lrow) * E_ + half * 8;
        uint4 bq[2][2];
        cl_issue(brow1, bq[0][0], bq[0][1]);
        #pragma unroll 2
        for (int i = 0; i < 64; ++i) {
            v16bf A = ldfrag(arow1 + i * 32);          // ds load before fence
            cl_issue(brow1 + ((i + 1) & 63) * 32,
                     bq[(i + 1) & 1][0], bq[(i + 1) & 1][1]);
            cl_wait<2>();                              // pair i retired
            c1 = wmma_pair(A, bq[i & 1][0], bq[i & 1][1], c1);
        }
        cl_drain();                       // retire the wrapped duplicate pair

        float bias1 = ldsb1[hc + nt1 * 16 + lrow];   // per column, from LDS
        #pragma unroll
        for (int r = 0; r < 8; ++r) {
            float v = c1[r] + bias1;
            v = 0.5f * v * (1.0f + erff(v * 0.70710678118654752f));  // exact GELU
            hbuf[(mt1 * 16 + r + 8 * half) * HROW + nt1 * 16 + lrow] = f2bf(v);
        }
        __syncthreads();                  // h-tile visible (single barrier/chunk)

        // ---- GEMM2: acc += h_tile * W2_chunk^T, grouped cluster-load pipeline
        #pragma unroll
        for (int kb = 0; kb < HC; kb += 32) {
            v16bf A0 = ldfrag(hbuf + lrow * HROW + kb + half * 8);
            v16bf A1 = ldfrag(hbuf + (16 + lrow) * HROW + kb + half * 8);
            const unsigned short* bbase =
                w2 + (size_t)(wcol0 + lrow) * H_ + hc + kb + half * 8;
            #pragma unroll
            for (int g = 0; g < 2; ++g) {           // two groups of 4 B-pairs
                uint4 q[4][2];
                const unsigned short* bg = bbase + (size_t)(g * 4) * 16 * H_;
                cl_issue(bg + (size_t)0 * 16 * H_, q[0][0], q[0][1]);
                cl_issue(bg + (size_t)1 * 16 * H_, q[1][0], q[1][1]);
                cl_issue(bg + (size_t)2 * 16 * H_, q[2][0], q[2][1]);
                cl_issue(bg + (size_t)3 * 16 * H_, q[3][0], q[3][1]);
                cl_wait<6>();
                acc[0][g*4+0] = wmma_pair(A0, q[0][0], q[0][1], acc[0][g*4+0]);
                acc[1][g*4+0] = wmma_pair(A1, q[0][0], q[0][1], acc[1][g*4+0]);
                cl_wait<4>();
                acc[0][g*4+1] = wmma_pair(A0, q[1][0], q[1][1], acc[0][g*4+1]);
                acc[1][g*4+1] = wmma_pair(A1, q[1][0], q[1][1], acc[1][g*4+1]);
                cl_wait<2>();
                acc[0][g*4+2] = wmma_pair(A0, q[2][0], q[2][1], acc[0][g*4+2]);
                acc[1][g*4+2] = wmma_pair(A1, q[2][0], q[2][1], acc[1][g*4+2]);
                cl_wait<0>();
                acc[0][g*4+3] = wmma_pair(A0, q[3][0], q[3][1], acc[0][g*4+3]);
                acc[1][g*4+3] = wmma_pair(A1, q[3][0], q[3][1], acc[1][g*4+3]);
            }
        }
        // loadcnt == 0 here; next chunk's accounting starts clean
    }

    // ---- Epilogue: + b2 (from LDS), store f32 ----
    #pragma unroll
    for (int mt = 0; mt < 2; ++mt) {
        #pragma unroll
        for (int nt = 0; nt < 8; ++nt) {
            int col = wcol0 + nt * 16 + lrow;
            float bias2 = ldsb2[col];
            #pragma unroll
            for (int r = 0; r < 8; ++r) {
                int gm = gm0 + mt * 16 + r + 8 * half;
                out[(size_t)gm * E_ + col] = acc[mt][nt][r] + bias2;
            }
        }
    }
}

extern "C" void kernel_launch(void* const* d_in, const int* in_sizes, int n_in,
                              void* d_out, int out_size, void* d_ws, size_t ws_size,
                              hipStream_t stream) {
    const float* x  = (const float*)d_in[0];
    const float* w1 = (const float*)d_in[1];   // proj1 [H, E]
    const float* b1 = (const float*)d_in[2];
    const float* w2 = (const float*)d_in[3];   // proj2 [E, H]
    const float* b2 = (const float*)d_in[4];
    // d_in[5] (gate_w) dead: routing is an identity permutation of shared weights.

    unsigned short* xb  = (unsigned short*)d_ws;             // [M*E] bf16
    unsigned short* w1b = xb  + (size_t)M_ * E_;             // [H*E] bf16
    unsigned short* w2b = w1b + (size_t)H_ * E_;             // [E*H] bf16

    moe_cvt_bf16<<<8192, 256, 0, stream>>>(x,  xb,  (M_ * E_) / 4);
    moe_cvt_bf16<<<4096, 256, 0, stream>>>(w1, w1b, (H_ * E_) / 4);
    moe_cvt_bf16<<<4096, 256, 0, stream>>>(w2, w2b, (H_ * E_) / 4);

    (void)hipFuncSetAttribute((const void*)moe_mlp_fused,
                              hipFuncAttributeMaxDynamicSharedMemorySize,
                              LDS_TOTAL);
    moe_mlp_fused<<<M_ / MT, TPB, LDS_TOTAL, stream>>>(xb, w1b, w2b, b1, b2,
                                                       (float*)d_out);
}
